// LatentLerpResidualInterpolator_82334523064718
// MI455X (gfx1250) — compile-verified, matrix operand order checked by reading, not run on hardware
//
#include <hip/hip_runtime.h>

typedef __attribute__((ext_vector_type(16))) __bf16 v16bf;
typedef __attribute__((ext_vector_type(8)))  float  v8f;
typedef int b128i __attribute__((vector_size(16)));  // matches async-LDS builtin param

#define NIMG 512   // B*T
#define HW   32
#define NPIX 1024

#if defined(__has_builtin)
#if __has_builtin(__builtin_amdgcn_global_load_async_to_lds_b128) && \
    __has_builtin(__builtin_amdgcn_s_wait_asynccnt)
#define HAVE_ASYNC 1
#endif
#endif
#ifndef HAVE_ASYNC
#define HAVE_ASYNC 0
#endif

// 16-byte global -> LDS copy: async-to-LDS path on CDNA5 (ASYNCcnt), else
// plain load/store fallback.
__device__ __forceinline__ void cp_b128(const uint4* __restrict__ g, uint4* l) {
#if HAVE_ASYNC
  __builtin_amdgcn_global_load_async_to_lds_b128(
      (__attribute__((address_space(1))) b128i*)g,
      (__attribute__((address_space(3))) b128i*)l, 0, 0);
#else
  *l = *g;
#endif
}

__device__ __forceinline__ void wait_async() {
#if HAVE_ASYNC
  __builtin_amdgcn_s_wait_asynccnt(0);
#endif
}

__device__ __forceinline__ float silu_f(float x) {
  return x * (1.0f / (1.0f + __expf(-x)));
}

__device__ __forceinline__ void calc_alpha(const int* __restrict__ idx, int b, int t,
                                           int& t0, int& t1, float& a, float& araw) {
  int seg = 0;
#pragma unroll
  for (int j = 1; j < 9; ++j)
    if (idx[b * 9 + j] <= t) seg = j;
  if (seg > 7) seg = 7;
  t0 = idx[b * 9 + seg];
  t1 = idx[b * 9 + seg + 1];
  int dt = t1 - t0;
  if (dt < 1) dt = 1;
  araw = (float)(t - t0) / (float)dt;
  a = araw;
  if (a < 0.f) a = 0.f;
  if (a > 1.f) a = 1.f;
}

// ---------------------------------------------------------------------------
// Weight prep: fp32 [cout][cin][3][3]  ->  bf16 [tap][coutP][cinP] (zero pad)
// ---------------------------------------------------------------------------
__global__ void prep_w(const float* __restrict__ src, __bf16* __restrict__ dst,
                       int cout, int cin, int coutP, int cinP) {
  int id = blockIdx.x * 256 + threadIdx.x;
  int tot = 9 * coutP * cinP;
  if (id >= tot) return;
  int tap = id / (coutP * cinP);
  int rem = id - tap * (coutP * cinP);
  int co = rem / cinP;
  int ci = rem - co * cinP;
  float v = 0.0f;
  if (co < cout && ci < cin) v = src[((size_t)co * cin + ci) * 9 + tap];
  dst[id] = (__bf16)v;
}

// ---------------------------------------------------------------------------
// Build res_in: channel-last bf16 [img][y][x][32]:
//   ch 0-7 z_base, 8-15 z0, 16-23 z1, 24 alpha, 25-31 zero
// ---------------------------------------------------------------------------
__global__ __launch_bounds__(256) void build_input(const float* __restrict__ latents,
                                                   const int* __restrict__ idx,
                                                   __bf16* __restrict__ resin) {
  const int img = blockIdx.x;
  const int b = img >> 7, t = img & 127;
  int t0, t1; float a, araw;
  calc_alpha(idx, b, t, t0, t1, a, araw);
  const float a1 = 1.0f - a;
  const float* L0 = latents + ((size_t)b * 128 + t0) * 8 * NPIX;
  const float* L1 = latents + ((size_t)b * 128 + t1) * 8 * NPIX;

  for (int p = threadIdx.x; p < NPIX; p += 256) {
    __align__(16) __bf16 buf[32];
#pragma unroll
    for (int c = 0; c < 8; ++c) {
      float z0 = L0[(size_t)c * NPIX + p];
      float z1 = L1[(size_t)c * NPIX + p];
      buf[c]      = (__bf16)(a1 * z0 + a * z1);
      buf[8 + c]  = (__bf16)z0;
      buf[16 + c] = (__bf16)z1;
    }
    buf[24] = (__bf16)a;
#pragma unroll
    for (int c = 25; c < 32; ++c) buf[c] = (__bf16)0.0f;
    uint4* dst = (uint4*)(resin + ((size_t)img * NPIX + p) * 32);
    const uint4* srcv = (const uint4*)buf;
#pragma unroll
    for (int i = 0; i < 4; ++i) dst[i] = srcv[i];
  }
}

// ---------------------------------------------------------------------------
// Implicit-GEMM 3x3 conv via WMMA bf16, async-staged LDS, per-tap
// double-buffered weights so the async copy overlaps the XDL work.
// in : [img][32][32][CINP] bf16, out: [img][32][32][64] bf16
// wgt: [9][64][CINP] bf16, bias fp32[64]
// Workgroup: 4 rows of one image; wave = 16-pixel M-tile x 64 outputs.
// ---------------------------------------------------------------------------
template <int CINP, bool SILU, bool RESID>
__global__ __launch_bounds__(256) void conv3x3_wmma(const __bf16* __restrict__ in,
                                                    __bf16* __restrict__ out,
                                                    const __bf16* __restrict__ wgt,
                                                    const float* __restrict__ bias,
                                                    const __bf16* __restrict__ resid) {
  constexpr int COUTP = 64;
  constexpr int NT = COUTP / 16;
  __shared__ __align__(32) __bf16 sAct[6 * 34 * CINP];     // rows y0-1..y0+4, cols -1..32
  __shared__ __align__(32) __bf16 sW[2][COUTP * CINP];     // per-tap double buffer

  const int img = blockIdx.y;
  const int y0 = blockIdx.x * 4;
  const int tid = threadIdx.x;

  // ---- stage activation tile (zero halo via ds_store, interior via async) ----
  {
    constexpr int perPix = CINP / 8;  // uint4 per pixel
    constexpr int total = 6 * 34 * perPix;
    uint4* dst = (uint4*)sAct;
    for (int i = tid; i < total; i += 256) {
      int pix = i / perPix, c4 = i - pix * perPix;
      int yy = pix / 34, xx = pix - yy * 34;
      int gy = y0 - 1 + yy, gx = xx - 1;
      if (gy >= 0 && gy < HW && gx >= 0 && gx < HW) {
        cp_b128(((const uint4*)(in + ((size_t)img * NPIX + gy * HW + gx) * CINP)) + c4,
                dst + i);
      } else {
        uint4 z = {0u, 0u, 0u, 0u};
        dst[i] = z;
      }
    }
  }
  // ---- stage weights for tap 0 ----
  constexpr int nqW = COUTP * CINP / 8;
  {
    const uint4* src = (const uint4*)wgt;
    uint4* dst = (uint4*)sW[0];
    for (int i = tid; i < nqW; i += 256) cp_b128(src + i, dst + i);
  }
  wait_async();
  __syncthreads();

  const int wave = tid >> 5;
  const int lane = tid & 31;
  const int row = wave >> 1;        // 0..3
  const int xb = (wave & 1) * 16;   // 0 or 16
  const int lm = lane & 15;
  const int kh = lane >> 4;

  v8f acc[NT];
#pragma unroll
  for (int nt = 0; nt < NT; ++nt)
#pragma unroll
    for (int j = 0; j < 8; ++j) acc[nt][j] = 0.0f;

  for (int tap = 0; tap < 9; ++tap) {
    // prefetch next tap's weights into the other buffer (overlaps WMMA below)
    if (tap < 8) {
      const uint4* src = (const uint4*)(wgt + (size_t)(tap + 1) * COUTP * CINP);
      uint4* dst = (uint4*)sW[(tap + 1) & 1];
      for (int i = tid; i < nqW; i += 256) cp_b128(src + i, dst + i);
    }
    const int dy = tap / 3 - 1;
    const int dx = tap % 3 - 1;
    const __bf16* aBase =
        &sAct[(size_t)((row + 1 + dy) * 34 + (xb + lm + 1 + dx)) * CINP + kh * 16];
    const __bf16* wBase = &sW[tap & 1][kh * 16];
#pragma unroll
    for (int cc = 0; cc < CINP / 32; ++cc) {
      v16bf afrag = *(const v16bf*)(aBase + cc * 32);
#pragma unroll
      for (int nt = 0; nt < NT; ++nt) {
        v16bf bfrag = *(const v16bf*)(wBase + (size_t)(nt * 16 + lm) * CINP + cc * 32);
        acc[nt] = __builtin_amdgcn_wmma_f32_16x16x32_bf16(
            false, afrag, false, bfrag, (short)0, acc[nt], false, false);
      }
    }
    wait_async();
    __syncthreads();
  }

  // ---- epilogue: bias, SiLU, optional residual (elementwise, in-place safe) ----
  const int gy = y0 + row;
#pragma unroll
  for (int nt = 0; nt < NT; ++nt) {
    const int n = nt * 16 + lm;
    const float bv = bias[n];
#pragma unroll
    for (int r = 0; r < 8; ++r) {
      const int gx = xb + r + kh * 8;
      float v = acc[nt][r] + bv;
      if (SILU) v = silu_f(v);
      size_t o = ((size_t)img * NPIX + gy * HW + gx) * 64 + n;
      if (RESID) v += (float)resid[o];
      out[o] = (__bf16)v;
    }
  }
}

// ---------------------------------------------------------------------------
// conv_out (64 -> 9, padded 16) fused with final lerp + confidence.
// ---------------------------------------------------------------------------
__global__ __launch_bounds__(256) void conv_out_final(const __bf16* __restrict__ in,
                                                      const __bf16* __restrict__ wgt,
                                                      const float* __restrict__ bias,
                                                      const float* __restrict__ latents,
                                                      const int* __restrict__ idx,
                                                      float* __restrict__ zout,
                                                      float* __restrict__ confout) {
  constexpr int CINP = 64;
  constexpr int COUTP = 16;
  __shared__ __align__(32) __bf16 sAct[6 * 34 * CINP];
  __shared__ __align__(32) __bf16 sW[2][COUTP * CINP];

  const int img = blockIdx.y;
  const int y0 = blockIdx.x * 4;
  const int tid = threadIdx.x;

  {
    constexpr int perPix = CINP / 8;
    constexpr int total = 6 * 34 * perPix;
    uint4* dst = (uint4*)sAct;
    for (int i = tid; i < total; i += 256) {
      int pix = i / perPix, c4 = i - pix * perPix;
      int yy = pix / 34, xx = pix - yy * 34;
      int gy = y0 - 1 + yy, gx = xx - 1;
      if (gy >= 0 && gy < HW && gx >= 0 && gx < HW) {
        cp_b128(((const uint4*)(in + ((size_t)img * NPIX + gy * HW + gx) * CINP)) + c4,
                dst + i);
      } else {
        uint4 z = {0u, 0u, 0u, 0u};
        dst[i] = z;
      }
    }
  }
  constexpr int nqW = COUTP * CINP / 8;
  {
    const uint4* src = (const uint4*)wgt;
    uint4* dst = (uint4*)sW[0];
    for (int i = tid; i < nqW; i += 256) cp_b128(src + i, dst + i);
  }
  wait_async();
  __syncthreads();

  const int wave = tid >> 5;
  const int lane = tid & 31;
  const int row = wave >> 1;
  const int xb = (wave & 1) * 16;
  const int lm = lane & 15;
  const int kh = lane >> 4;

  v8f acc;
#pragma unroll
  for (int j = 0; j < 8; ++j) acc[j] = 0.0f;

  for (int tap = 0; tap < 9; ++tap) {
    if (tap < 8) {
      const uint4* src = (const uint4*)(wgt + (size_t)(tap + 1) * COUTP * CINP);
      uint4* dst = (uint4*)sW[(tap + 1) & 1];
      for (int i = tid; i < nqW; i += 256) cp_b128(src + i, dst + i);
    }
    const int dy = tap / 3 - 1;
    const int dx = tap % 3 - 1;
    const __bf16* aBase =
        &sAct[(size_t)((row + 1 + dy) * 34 + (xb + lm + 1 + dx)) * CINP + kh * 16];
    const __bf16* wBase = &sW[tap & 1][(size_t)lm * CINP + kh * 16];
#pragma unroll
    for (int cc = 0; cc < CINP / 32; ++cc) {
      v16bf afrag = *(const v16bf*)(aBase + cc * 32);
      v16bf bfrag = *(const v16bf*)(wBase + cc * 32);
      acc = __builtin_amdgcn_wmma_f32_16x16x32_bf16(
          false, afrag, false, bfrag, (short)0, acc, false, false);
    }
    wait_async();
    __syncthreads();
  }

  // ---- fused finalize ----
  const int b = img >> 7, t = img & 127;
  int t0, t1; float a, araw;
  calc_alpha(idx, b, t, t0, t1, a, araw);
  const bool interior = (araw > 0.0f) && (araw < 1.0f);
  const float s = a * (1.0f - a);
  const float a1 = 1.0f - a;
  const int n = lm;  // output channel (0..8 valid)
  const float bv = (n < 9) ? bias[n] : 0.0f;
  const int gy = y0 + row;

#pragma unroll
  for (int r = 0; r < 8; ++r) {
    const int gx = xb + r + kh * 8;
    const float v = acc[r] + bv;
    const size_t p = (size_t)gy * HW + gx;
    if (n < 8) {
      float z0 = latents[(((size_t)b * 128 + t0) * 8 + n) * NPIX + p];
      float z1 = latents[(((size_t)b * 128 + t1) * 8 + n) * NPIX + p];
      float zb = a1 * z0 + a * z1;
      zout[(((size_t)b * 128 + t) * 8 + n) * NPIX + p] = zb + s * v;
    } else if (n == 8) {
      float unc = 1.0f / (1.0f + __expf(-v));
      float c = interior ? (1.0f - unc) : 1.0f;
      if (c < 0.0f) c = 0.0f;
      if (c > 1.0f) c = 1.0f;
      confout[((size_t)b * 128 + t) * NPIX + p] = c;
    }
  }
}

// ---------------------------------------------------------------------------
extern "C" void kernel_launch(void* const* d_in, const int* in_sizes, int n_in,
                              void* d_out, int out_size, void* d_ws, size_t ws_size,
                              hipStream_t stream) {
  (void)in_sizes; (void)n_in; (void)out_size; (void)ws_size;
  const float* latents = (const float*)d_in[0];
  const int*   idx     = (const int*)d_in[1];
  const float* w_in    = (const float*)d_in[2];
  const float* b_in    = (const float*)d_in[3];
  const float* w1s     = (const float*)d_in[4];
  const float* b1s     = (const float*)d_in[5];
  const float* w2s     = (const float*)d_in[6];
  const float* b2s     = (const float*)d_in[7];
  const float* w_out   = (const float*)d_in[8];
  const float* b_out   = (const float*)d_in[9];

  char* ws = (char*)d_ws;
  size_t off = 0;
  auto take = [&](size_t elems) {
    __bf16* p = (__bf16*)(ws + off);
    off += ((elems * 2 + 255) / 256) * 256;
    return p;
  };
  __bf16* wIn   = take((size_t)9 * 64 * 32);
  __bf16* wB0a  = take((size_t)9 * 64 * 64);
  __bf16* wB0b  = take((size_t)9 * 64 * 64);
  __bf16* wB1a  = take((size_t)9 * 64 * 64);
  __bf16* wB1b  = take((size_t)9 * 64 * 64);
  __bf16* wOut  = take((size_t)9 * 16 * 64);
  __bf16* resin = take((size_t)NIMG * NPIX * 32);
  __bf16* actA  = take((size_t)NIMG * NPIX * 64);
  __bf16* actB  = take((size_t)NIMG * NPIX * 64);

  auto prep = [&](const float* src, __bf16* dst, int cout, int cin, int coutP, int cinP) {
    int tot = 9 * coutP * cinP;
    prep_w<<<dim3((tot + 255) / 256), dim3(256), 0, stream>>>(src, dst, cout, cin, coutP, cinP);
  };
  prep(w_in, wIn, 64, 25, 64, 32);
  prep(w1s + (size_t)0 * 64 * 64 * 9, wB0a, 64, 64, 64, 64);
  prep(w2s + (size_t)0 * 64 * 64 * 9, wB0b, 64, 64, 64, 64);
  prep(w1s + (size_t)1 * 64 * 64 * 9, wB1a, 64, 64, 64, 64);
  prep(w2s + (size_t)1 * 64 * 64 * 9, wB1b, 64, 64, 64, 64);
  prep(w_out, wOut, 9, 64, 16, 64);

  build_input<<<dim3(NIMG), dim3(256), 0, stream>>>(latents, idx, resin);

  dim3 grid(8, NIMG);
  dim3 blk(256);
  conv3x3_wmma<32, true,  false><<<grid, blk, 0, stream>>>(resin, actA, wIn, b_in, nullptr);
  conv3x3_wmma<64, true,  false><<<grid, blk, 0, stream>>>(actA, actB, wB0a, b1s, nullptr);
  conv3x3_wmma<64, false, true ><<<grid, blk, 0, stream>>>(actB, actA, wB0b, b2s, actA);
  conv3x3_wmma<64, true,  false><<<grid, blk, 0, stream>>>(actA, actB, wB1a, b1s + 64, nullptr);
  conv3x3_wmma<64, false, true ><<<grid, blk, 0, stream>>>(actB, actA, wB1b, b2s + 64, actA);

  float* zout = (float*)d_out;
  float* confout = zout + (size_t)4 * 128 * 8 * 32 * 32;
  conv_out_final<<<grid, blk, 0, stream>>>(actA, wOut, b_out, latents, idx, zout, confout);
}
